// SparseMoELanguageModel_4569845203474
// MI455X (gfx1250) — compile-verified, matrix operand order checked by reading, not run on hardware
//
#include <hip/hip_runtime.h>
#include <math.h>

// ---------------- model dims ----------------
constexpr int Bb  = 8;
constexpr int Tt  = 512;
constexpr int Dd  = 1024;
constexpr int Hh  = 16;
constexpr int Ll  = 2;
constexpr int Ee  = 8;
constexpr int Kk  = 2;
constexpr int Fff = 4096;
constexpr int Ntok = Bb * Tt;          // 4096
constexpr int CAPc = Ntok * Kk / Ee;   // 1024
constexpr int HDh  = Dd / Hh;          // 64

// ---------------- vector types ----------------
typedef __bf16 v16bf __attribute__((ext_vector_type(16)));
typedef __bf16 v8bf  __attribute__((ext_vector_type(8)));
typedef float  v8f   __attribute__((ext_vector_type(8)));

// ---------------- helpers ----------------
__device__ __forceinline__ unsigned short f2bf(float f) {
    union { float f; unsigned u; } x; x.f = f;
    unsigned u = x.u;
    unsigned r = u + 0x7FFFu + ((u >> 16) & 1u);   // round-to-nearest-even
    return (unsigned short)(r >> 16);
}
__device__ __forceinline__ float bf2f(unsigned short h) {
    union { float f; unsigned u; } x; x.u = ((unsigned)h) << 16;
    return x.f;
}
__device__ __forceinline__ float waveSum(float v) {
    #pragma unroll
    for (int o = 16; o > 0; o >>= 1) v += __shfl_xor(v, o, 32);
    return v;
}
__device__ __forceinline__ float waveMax(float v) {
    #pragma unroll
    for (int o = 16; o > 0; o >>= 1) v = fmaxf(v, __shfl_xor(v, o, 32));
    return v;
}

// ================= generic batched bf16 WMMA GEMM =================
// C[z][m][n] = alpha * sum_k A[z][m][k] * B[z][k][n] (+bias[n]) (ReLU?)
// A bf16 row-major (lda). B bf16: TRANS_B==0 -> B[k*ldb+n], TRANS_B==1 -> B[n*ldb+k].
// Batch z decomposed: zo = z / innerCount, zi = z % innerCount.
// Block tile: 128 x BNt x 32. 8 waves in a 4(M) x 2(N) grid; each wave owns a
// 32 x (BNt/2) tile = 2 x (BNt/32) WMMA fragments with fp32 accumulation.
#define BM 128
#define BK 32
#define LDS_STRIDE (BK + 8)   // 40 halfwords -> skewed banks, 16B-aligned fragments

template<int BNt, bool TRANS_B, bool BIAS, bool RELU, bool OUT_BF16>
__global__ __launch_bounds__(256)
void gemm_wmma(const unsigned short* __restrict__ Aall,
               const unsigned short* __restrict__ Ball,
               void* __restrict__ Call,
               const float* __restrict__ biasAll,
               int M, int Nn, int Kd,
               int lda, int ldb, int ldc,
               long long sAo, long long sAi,
               long long sBo, long long sBi,
               long long sCo, long long sCi,
               long long sBias, int innerCount, float alpha)
{
    constexpr int FN = BNt / 32;               // N-fragments per wave
    constexpr int AV = (BM * BK) / (256 * 8);  // uint4 loads/thread for A
    constexpr int BV = (BNt * BK) / (256 * 8); // uint4 loads/thread for B

    __shared__ unsigned short lA[BM][LDS_STRIDE];
    __shared__ unsigned short lB[BNt][LDS_STRIDE];

    const int z  = blockIdx.z;
    const int zo = z / innerCount;
    const int zi = z - zo * innerCount;
    const unsigned short* A = Aall + zo * sAo + zi * sAi;
    const unsigned short* B = Ball + zo * sBo + zi * sBi;
    const long long coff = zo * sCo + zi * sCi;
    const float* bias = nullptr;
    if (BIAS) bias = biasAll + (long long)z * sBias;

    const int tid   = threadIdx.x;
    const int lane  = tid & 31;
    const int wave  = tid >> 5;
    const int waveM = wave >> 1;     // 0..3  (M direction)
    const int waveN = wave & 1;      // 0..1  (N direction)
    const int half  = lane >> 4;     // 0/1
    const int l16   = lane & 15;

    const int m0 = blockIdx.y * BM;
    const int n0 = blockIdx.x * BNt;

    v8f acc[2][FN] = {};

    for (int k0 = 0; k0 < Kd; k0 += BK) {
        // ---- stage A tile (BM x 32), b128 all the way ----
        #pragma unroll
        for (int v = 0; v < AV; ++v) {
            int idx = tid + v * 256;
            int row = idx >> 2;            // BK/8 = 4 vectors per row
            int col = (idx & 3) * 8;
            uint4 a = *reinterpret_cast<const uint4*>(
                A + (long long)(m0 + row) * lda + k0 + col);
            *reinterpret_cast<uint4*>(&lA[row][col]) = a;
        }
        // ---- stage B tile, LDS layout [n][k] (BNt x 32) ----
        if (TRANS_B) {
            // B already k-contiguous per n: pure b128 path (hot path)
            #pragma unroll
            for (int v = 0; v < BV; ++v) {
                int idx = tid + v * 256;
                int row = idx >> 2;
                int col = (idx & 3) * 8;
                uint4 b = *reinterpret_cast<const uint4*>(
                    B + (long long)(n0 + row) * ldb + k0 + col);
                *reinterpret_cast<uint4*>(&lB[row][col]) = b;
            }
        } else {
            // n-contiguous source: scalar transpose into LDS (only small P@V GEMM)
            constexpr int perRow = BNt / 8;
            #pragma unroll
            for (int v = 0; v < BV; ++v) {
                int idx = tid + v * 256;
                int k  = idx / perRow;
                int nc = (idx % perRow) * 8;
                uint4 b = *reinterpret_cast<const uint4*>(
                    B + (long long)(k0 + k) * ldb + n0 + nc);
                const unsigned short* e = reinterpret_cast<const unsigned short*>(&b);
                #pragma unroll
                for (int i = 0; i < 8; ++i) lB[nc + i][k] = e[i];
            }
        }
        // prefetch next K tile (global_prefetch_b8)
        if (k0 + BK < Kd) {
            __builtin_prefetch(A + (long long)(m0 + (tid >> 2)) * lda + (k0 + BK), 0, 3);
            if (TRANS_B)
                __builtin_prefetch(B + (long long)(n0 + (tid >> 2)) * ldb + (k0 + BK), 0, 3);
            else
                __builtin_prefetch(B + (long long)(k0 + BK + (tid >> 3)) * ldb + n0, 0, 3);
        }
        __syncthreads();

        // ---- load fragments per CDNA5 16-bit A/B VGPR layout ----
        v16bf af[2], bfr[FN];
        #pragma unroll
        for (int fm = 0; fm < 2; ++fm) {
            int m = waveM * 32 + fm * 16 + l16;
            v8bf lo = *reinterpret_cast<const v8bf*>(&lA[m][half * 8]);
            v8bf hi = *reinterpret_cast<const v8bf*>(&lA[m][16 + half * 8]);
            af[fm] = __builtin_shufflevector(lo, hi, 0,1,2,3,4,5,6,7,8,9,10,11,12,13,14,15);
        }
        #pragma unroll
        for (int fn = 0; fn < FN; ++fn) {
            int n = waveN * (BNt / 2) + fn * 16 + l16;
            v8bf lo = *reinterpret_cast<const v8bf*>(&lB[n][half * 8]);
            v8bf hi = *reinterpret_cast<const v8bf*>(&lB[n][16 + half * 8]);
            bfr[fn] = __builtin_shufflevector(lo, hi, 0,1,2,3,4,5,6,7,8,9,10,11,12,13,14,15);
        }
        // ---- 2*FN WMMAs per wave per K step ----
        #pragma unroll
        for (int fm = 0; fm < 2; ++fm)
            #pragma unroll
            for (int fn = 0; fn < FN; ++fn)
                acc[fm][fn] = __builtin_amdgcn_wmma_f32_16x16x32_bf16(
                    false, af[fm], false, bfr[fn], (short)0, acc[fm][fn], false, false);
        __syncthreads();
    }

    // ---- epilogue: C/D layout m = r + 8*half, n = l16 ----
    #pragma unroll
    for (int fm = 0; fm < 2; ++fm) {
        #pragma unroll
        for (int fn = 0; fn < FN; ++fn) {
            int n = n0 + waveN * (BNt / 2) + fn * 16 + l16;
            #pragma unroll
            for (int r = 0; r < 8; ++r) {
                int m = m0 + waveM * 32 + fm * 16 + r + 8 * half;
                float v = acc[fm][fn][r] * alpha;
                if (BIAS) v += bias[n];
                if (RELU) v = v > 0.f ? v : 0.f;
                long long ci = coff + (long long)m * ldc + n;
                if (OUT_BF16) reinterpret_cast<unsigned short*>(Call)[ci] = f2bf(v);
                else          reinterpret_cast<float*>(Call)[ci] = v;
            }
        }
    }
}

// ================= fp32 -> bf16 transposed weight conversion =================
// in: fp32 [R][C] row-major; out: bf16 [C][R] row-major. Batched via blockIdx.z.
__global__ __launch_bounds__(256)
void cvt_T_k(const float* __restrict__ in, unsigned short* __restrict__ out,
             int R, int C)
{
    __shared__ float t[32][33];
    long long base = (long long)blockIdx.z * R * C;
    int r0 = blockIdx.y * 32, c0 = blockIdx.x * 32;
    int tx = threadIdx.x & 31, ty = threadIdx.x >> 5;
    #pragma unroll
    for (int i = 0; i < 4; ++i)
        t[ty + 8 * i][tx] = in[base + (long long)(r0 + ty + 8 * i) * C + c0 + tx];
    __syncthreads();
    #pragma unroll
    for (int i = 0; i < 4; ++i)
        out[base + (long long)(c0 + ty + 8 * i) * R + r0 + tx] = f2bf(t[tx][ty + 8 * i]);
}

// ================= embedding =================
__global__ __launch_bounds__(256)
void embed_k(const int* __restrict__ ids, const float* __restrict__ tok,
             const float* __restrict__ pos, float* __restrict__ x)
{
    int n = blockIdx.x;
    int t = n % Tt;
    long long tb = (long long)ids[n] * Dd;
    #pragma unroll
    for (int i = 0; i < 4; ++i) {
        int d = threadIdx.x + i * 256;
        x[(long long)n * Dd + d] = tok[tb + d] + pos[(long long)t * Dd + d];
    }
}

// ================= layernorm (D=1024, one row per block) =================
template<bool OUT_BF16>
__global__ __launch_bounds__(256)
void layernorm_k(const float* __restrict__ X, const float* __restrict__ g,
                 const float* __restrict__ b, void* __restrict__ Y)
{
    long long row = blockIdx.x;
    const float* x = X + row * Dd;
    int tid = threadIdx.x;
    float vals[4], s = 0.f, sq = 0.f;
    #pragma unroll
    for (int i = 0; i < 4; ++i) {
        float v = x[tid + i * 256];
        vals[i] = v; s += v; sq += v * v;
    }
    __shared__ float s1[8], s2[8];
    float ws = waveSum(s), wq = waveSum(sq);
    if ((tid & 31) == 0) { s1[tid >> 5] = ws; s2[tid >> 5] = wq; }
    __syncthreads();
    float tot = 0.f, totq = 0.f;
    #pragma unroll
    for (int i = 0; i < 8; ++i) { tot += s1[i]; totq += s2[i]; }
    float mean = tot / Dd;
    float var  = totq / Dd - mean * mean;
    float rstd = rsqrtf(var + 1e-5f);
    #pragma unroll
    for (int i = 0; i < 4; ++i) {
        int d = tid + i * 256;
        float o = (vals[i] - mean) * rstd * g[d] + b[d];
        if (OUT_BF16) reinterpret_cast<unsigned short*>(Y)[row * Dd + d] = f2bf(o);
        else          reinterpret_cast<float*>(Y)[row * Dd + d] = o;
    }
}

// ================= softmax over T=512 (one row per block) =================
__global__ __launch_bounds__(256)
void softmax_k(const float* __restrict__ S, unsigned short* __restrict__ P)
{
    long long row = blockIdx.x;
    const float* s = S + row * Tt;
    unsigned short* p = P + row * Tt;
    int tid = threadIdx.x;
    float v0 = s[tid], v1 = s[tid + 256];
    __shared__ float sm[8];
    float wm = waveMax(fmaxf(v0, v1));
    if ((tid & 31) == 0) sm[tid >> 5] = wm;
    __syncthreads();
    float gm = sm[0];
    #pragma unroll
    for (int i = 1; i < 8; ++i) gm = fmaxf(gm, sm[i]);
    float e0 = __expf(v0 - gm), e1 = __expf(v1 - gm);
    __syncthreads();
    float wsum = waveSum(e0 + e1);
    if ((tid & 31) == 0) sm[tid >> 5] = wsum;
    __syncthreads();
    float gs = 0.f;
    #pragma unroll
    for (int i = 0; i < 8; ++i) gs += sm[i];
    float inv = 1.f / gs;
    p[tid]       = f2bf(e0 * inv);
    p[tid + 256] = f2bf(e1 * inv);
}

// ================= misc elementwise =================
__global__ void add_k(float* __restrict__ x, const float* __restrict__ y, long long n)
{
    long long i = (long long)blockIdx.x * 256 + threadIdx.x;
    long long stride = (long long)gridDim.x * 256;
    for (; i < n; i += stride) x[i] += y[i];
}

// ================= router: logits, noisy top-2 gating =================
__global__ __launch_bounds__(256)
void router_k(const unsigned short* __restrict__ xln,
              const float* __restrict__ rt_w, const float* __restrict__ rt_b,
              const float* __restrict__ nz_w, const float* __restrict__ nz_b,
              const float* __restrict__ noise,
              int* __restrict__ topi, float* __restrict__ topg)
{
    int n = blockIdx.x;
    int tid = threadIdx.x, w = tid >> 5, lane = tid & 31;
    const unsigned short* xr = xln + (long long)n * Dd;
    float aL = 0.f, aN = 0.f;
    for (int k = lane; k < Dd; k += 32) {
        float xv = bf2f(xr[k]);
        aL += xv * rt_w[k * Ee + w];
        aN += xv * nz_w[k * Ee + w];
    }
    aL = waveSum(aL); aN = waveSum(aN);
    __shared__ float lg[8], nzv[8];
    if (lane == 0) { lg[w] = aL; nzv[w] = aN; }
    __syncthreads();
    if (tid == 0) {
        float noisy[8];
        #pragma unroll
        for (int e = 0; e < 8; ++e) {
            float spi = nzv[e] + nz_b[e];
            float sp  = spi > 20.f ? spi : log1pf(__expf(spi));
            noisy[e]  = lg[e] + rt_b[e] + noise[(long long)n * Ee + e] * sp;
        }
        int i1 = 0;
        #pragma unroll
        for (int e = 1; e < 8; ++e) if (noisy[e] > noisy[i1]) i1 = e;
        int i2 = -1;
        #pragma unroll
        for (int e = 0; e < 8; ++e)
            if (e != i1 && (i2 < 0 || noisy[e] > noisy[i2])) i2 = e;
        float v1 = noisy[i1], v2 = noisy[i2];
        float p2 = 1.f / (1.f + __expf(v1 - v2));
        float p1 = 1.f - p2;
        topi[n * 2] = i1; topi[n * 2 + 1] = i2;
        topg[n * 2] = p1; topg[n * 2 + 1] = p2;
    }
}

// ================= capacity dispatch: wave32 ballot prefix-scan =================
__global__ __launch_bounds__(32)
void dispatch_k(const int* __restrict__ topi, const float* __restrict__ topg,
                int* __restrict__ sel, float* __restrict__ gsel)
{
    int e = blockIdx.x;
    int lane = threadIdx.x;
    int count = 0;
    for (int base = 0; base < Ntok; base += 32) {
        int n = base + lane;
        int i0 = topi[n * 2], i1 = topi[n * 2 + 1];
        bool pred = (i0 == e) || (i1 == e);
        unsigned long long bm = __ballot(pred);
        unsigned mask = (unsigned)bm;
        int rank = count + __popc(mask & ((1u << lane) - 1u));
        if (pred && rank < CAPc) {
            sel[e * CAPc + rank]  = n;
            gsel[e * CAPc + rank] = (i0 == e) ? topg[n * 2] : topg[n * 2 + 1];
        }
        count += __popc(mask);
    }
    int c = count < CAPc ? count : CAPc;
    for (int i = c + lane; i < CAPc; i += 32) {
        sel[e * CAPc + i]  = -1;
        gsel[e * CAPc + i] = 0.f;
    }
}

// ================= gather / scatter =================
__global__ __launch_bounds__(256)
void gather_k(const int* __restrict__ sel, const unsigned short* __restrict__ xln,
              unsigned short* __restrict__ xe)
{
    int row = blockIdx.x;
    int n = sel[row];
    #pragma unroll
    for (int i = 0; i < 4; ++i) {
        int d = threadIdx.x + i * 256;
        xe[(long long)row * Dd + d] = (n >= 0) ? xln[(long long)n * Dd + d] : (unsigned short)0;
    }
}
__global__ __launch_bounds__(256)
void scatter_k(const int* __restrict__ sel, const float* __restrict__ gsel,
               const float* __restrict__ eo, float* __restrict__ x)
{
    int row = blockIdx.x;
    int n = sel[row];
    if (n < 0) return;
    float g = gsel[row];
    #pragma unroll
    for (int i = 0; i < 4; ++i) {
        int d = threadIdx.x + i * 256;
        atomicAdd(&x[(long long)n * Dd + d], eo[(long long)row * Dd + d] * g);
    }
}

// ================= host orchestration =================
extern "C" void kernel_launch(void* const* d_in, const int* in_sizes, int n_in,
                              void* d_out, int out_size, void* d_ws, size_t ws_size,
                              hipStream_t stream)
{
    (void)in_sizes; (void)n_in; (void)out_size; (void)ws_size;
    const int*   input_ids = (const int*)  d_in[0];
    const float* base_noise= (const float*)d_in[1];
    const float* tok_emb   = (const float*)d_in[2];
    const float* pos_emb   = (const float*)d_in[3];
    const float* ln1_g     = (const float*)d_in[4];
    const float* ln1_b     = (const float*)d_in[5];
    const float* ln2_g     = (const float*)d_in[6];
    const float* ln2_b     = (const float*)d_in[7];
    const float* qkv_w     = (const float*)d_in[8];
    const float* out_w     = (const float*)d_in[9];
    const float* rt_w      = (const float*)d_in[10];
    const float* rt_b      = (const float*)d_in[11];
    const float* nz_w      = (const float*)d_in[12];
    const float* nz_b      = (const float*)d_in[13];
    const float* e_w1      = (const float*)d_in[14];
    const float* e_b1      = (const float*)d_in[15];
    const float* e_w2      = (const float*)d_in[16];
    const float* e_b2      = (const float*)d_in[17];
    const float* lnf_g     = (const float*)d_in[18];
    const float* lnf_b     = (const float*)d_in[19];

    // workspace carve-out (256B aligned)
    char* w = (char*)d_ws;
    auto alloc = [&](size_t bytes) -> void* {
        void* p = (void*)w;
        w += (bytes + 255) & ~(size_t)255;
        return p;
    };
    float*          x      = (float*)         alloc((size_t)Ntok * Dd * 4);
    unsigned short* xln    = (unsigned short*)alloc((size_t)Ntok * Dd * 2);
    unsigned short* qkvbf  = (unsigned short*)alloc((size_t)Ntok * 3 * Dd * 2);
    float*          Sbuf   = (float*)         alloc((size_t)Bb * Hh * Tt * Tt * 4);
    unsigned short* Pbuf   = (unsigned short*)alloc((size_t)Bb * Hh * Tt * Tt * 2);
    unsigned short* attnbf = (unsigned short*)alloc((size_t)Ntok * Dd * 2);
    float*          tmp    = (float*)         alloc((size_t)Ntok * Dd * 4);
    unsigned short* wqkvT  = (unsigned short*)alloc((size_t)Dd * 3 * Dd * 2);   // [3D][D]
    unsigned short* woutT  = (unsigned short*)alloc((size_t)Dd * Dd * 2);       // [D][D]
    unsigned short* w1T    = (unsigned short*)alloc((size_t)Ee * Dd * Fff * 2); // [e][F][D]
    unsigned short* w2T    = (unsigned short*)alloc((size_t)Ee * Fff * Dd * 2); // [e][D][F]
    unsigned short* xe     = (unsigned short*)alloc((size_t)Ee * CAPc * Dd * 2);
    unsigned short* hbuf   = (unsigned short*)alloc((size_t)Ee * CAPc * Fff * 2);
    float*          eobuf  = (float*)         alloc((size_t)Ee * CAPc * Dd * 4);
    int*            topi   = (int*)           alloc((size_t)Ntok * 2 * 4);
    float*          topg   = (float*)         alloc((size_t)Ntok * 2 * 4);
    int*            sel    = (int*)           alloc((size_t)Ee * CAPc * 4);
    float*          gsel   = (float*)         alloc((size_t)Ee * CAPc * 4);

    const dim3 blk(256);

    // x = tok_emb[ids] + pos_emb
    embed_k<<<Ntok, blk, 0, stream>>>(input_ids, tok_emb, pos_emb, x);

    for (int l = 0; l < Ll; ++l) {
        // ---------- attention ----------
        // Wqkv^T : [D][3D] -> bf16 [3D][D]
        cvt_T_k<<<dim3(3 * Dd / 32, Dd / 32, 1), blk, 0, stream>>>(
            qkv_w + (size_t)l * Dd * 3 * Dd, wqkvT, Dd, 3 * Dd);
        layernorm_k<true><<<Ntok, blk, 0, stream>>>(x, ln1_g + l * Dd, ln1_b + l * Dd, xln);

        // qkv = xln @ Wqkv  (4096 x 3072 x 1024) -> bf16
        gemm_wmma<128, true, false, false, true>
            <<<dim3(3 * Dd / 128, Ntok / BM, 1), blk, 0, stream>>>(
            xln, wqkvT, qkvbf, nullptr,
            Ntok, 3 * Dd, Dd, Dd, Dd, 3 * Dd,
            0, 0, 0, 0, 0, 0, 0, 1, 1.f);

        // S[b,h] = Q K^T / sqrt(HD)   (512 x 512 x 64, batch 128)
        gemm_wmma<128, true, false, false, false>
            <<<dim3(Tt / 128, Tt / BM, Bb * Hh), blk, 0, stream>>>(
            qkvbf, qkvbf + Dd, Sbuf, nullptr,
            Tt, Tt, HDh, 3 * Dd, 3 * Dd, Tt,
            (long long)Tt * 3 * Dd, HDh,
            (long long)Tt * 3 * Dd, HDh,
            (long long)Hh * Tt * Tt, (long long)Tt * Tt,
            0, Hh, 0.125f);

        // P = softmax(S) -> bf16
        softmax_k<<<Bb * Hh * Tt, blk, 0, stream>>>(Sbuf, Pbuf);

        // O[b,h] = P @ V   (512 x 64 x 512, batch 128) -> bf16 into [b,t,h,hd]
        gemm_wmma<64, false, false, false, true>
            <<<dim3(HDh / 64, Tt / BM, Bb * Hh), blk, 0, stream>>>(
            Pbuf, qkvbf + 2 * Dd, attnbf, nullptr,
            Tt, HDh, Tt, Tt, 3 * Dd, Dd,
            (long long)Hh * Tt * Tt, (long long)Tt * Tt,
            (long long)Tt * 3 * Dd, HDh,
            (long long)Tt * Dd, HDh,
            0, Hh, 1.f);

        // proj = O @ Wout   (4096 x 1024 x 1024)
        cvt_T_k<<<dim3(Dd / 32, Dd / 32, 1), blk, 0, stream>>>(
            out_w + (size_t)l * Dd * Dd, woutT, Dd, Dd);
        gemm_wmma<128, true, false, false, false>
            <<<dim3(Dd / 128, Ntok / BM, 1), blk, 0, stream>>>(
            attnbf, woutT, tmp, nullptr,
            Ntok, Dd, Dd, Dd, Dd, Dd,
            0, 0, 0, 0, 0, 0, 0, 1, 1.f);
        add_k<<<2048, blk, 0, stream>>>(x, tmp, (long long)Ntok * Dd);

        // ---------- MoE ----------
        layernorm_k<true><<<Ntok, blk, 0, stream>>>(x, ln2_g + l * Dd, ln2_b + l * Dd, xln);
        router_k<<<Ntok, blk, 0, stream>>>(xln,
            rt_w + (size_t)l * Dd * Ee, rt_b + l * Ee,
            nz_w + (size_t)l * Dd * Ee, nz_b + l * Ee,
            base_noise + (size_t)l * Ntok * Ee, topi, topg);
        dispatch_k<<<Ee, dim3(32), 0, stream>>>(topi, topg, sel, gsel);
        gather_k<<<Ee * CAPc, blk, 0, stream>>>(sel, xln, xe);

        // W1^T : [e][D][F] -> bf16 [e][F][D]
        cvt_T_k<<<dim3(Fff / 32, Dd / 32, Ee), blk, 0, stream>>>(
            e_w1 + (size_t)l * Ee * Dd * Fff, w1T, Dd, Fff);
        // h = relu(xe @ W1 + b1)   (1024 x 4096 x 1024, batch 8) -> bf16
        gemm_wmma<128, true, true, true, true>
            <<<dim3(Fff / 128, CAPc / BM, Ee), blk, 0, stream>>>(
            xe, w1T, hbuf, e_b1 + (size_t)l * Ee * Fff,
            CAPc, Fff, Dd, Dd, Dd, Fff,
            (long long)CAPc * Dd, 0,
            (long long)Dd * Fff, 0,
            (long long)CAPc * Fff, 0,
            Fff, 1, 1.f);

        // W2^T : [e][F][D] -> bf16 [e][D][F]
        cvt_T_k<<<dim3(Dd / 32, Fff / 32, Ee), blk, 0, stream>>>(
            e_w2 + (size_t)l * Ee * Fff * Dd, w2T, Fff, Dd);
        // eo = h @ W2 + b2   (1024 x 1024 x 4096, batch 8) -> f32
        gemm_wmma<128, true, true, false, false>
            <<<dim3(Dd / 128, CAPc / BM, Ee), blk, 0, stream>>>(
            hbuf, w2T, eobuf, e_b2 + (size_t)l * Ee * Dd,
            CAPc, Dd, Fff, Fff, Fff, Dd,
            (long long)CAPc * Fff, 0,
            (long long)Fff * Dd, 0,
            (long long)CAPc * Dd, 0,
            Dd, 1, 1.f);

        // x += gate * eo (scatter-add, K=2 so tokens may collide -> atomics)
        scatter_k<<<Ee * CAPc, blk, 0, stream>>>(sel, gsel, eobuf, x);
    }

    // final layernorm -> fp32 output
    layernorm_k<false><<<Ntok, blk, 0, stream>>>(x, lnf_g, lnf_b, d_out);
}
